// PolicyHead_38036230374146
// MI455X (gfx1250) — matrix operand order repaired and checked by reading, not compiled, not run on hardware
//
#include <hip/hip_runtime.h>
#include <hip/hip_bf16.h>

typedef __attribute__((ext_vector_type(16))) _Float16 v16h;
typedef __attribute__((ext_vector_type(8)))  _Float16 v8h;
typedef __attribute__((ext_vector_type(8)))  float    v8f;
typedef __attribute__((ext_vector_type(4)))  float    v4f;
typedef __attribute__((ext_vector_type(4)))  unsigned int v4u;
typedef __attribute__((ext_vector_type(8)))  int      v8i;
typedef __attribute__((ext_vector_type(4)))  int      v4i;

#define WMMA_F16(a, b, c) \
  __builtin_amdgcn_wmma_f32_16x16x32_f16(false, (a), false, (b), (short)0, (c), false, false)

// ---- Workspace layout (halves) ----
// [0          , 131072) : kv_w^T  [512][256]  (B-operand layout, [N][K])
// [131072     , 196608) : out_w^T [256][256]
// [196608     , 217088) : pol_w^T [80][256]   (cols 73..79 zero-padded)
// [217088     , 233472) : q       [64][256]   (attn_map@q_w + q_b, pre-scaled)
#define WS_KVW  0
#define WS_OWT  131072
#define WS_POLT 196608
#define WS_Q    217088
#define WS_TOTAL 233472

// A fragment (16x32 f16), row-major f16 source, stride in halves.
// lane l: m = l&15; k runs [base, base+8) and [base+16, base+24), base = (l&16)>>1.
__device__ __forceinline__ v16h load_a_h(const _Float16* base, int stride, int lane) {
  int m  = lane & 15;
  int kb = (lane & 16) >> 1;
  const _Float16* p = base + m * stride + kb;
  v8h lo = *(const v8h*)(p);
  v8h hi = *(const v8h*)(p + 16);
  return __builtin_shufflevector(lo, hi, 0,1,2,3,4,5,6,7,8,9,10,11,12,13,14,15);
}

// Same A fragment but from f32 source (vectorized loads, convert on the fly).
__device__ __forceinline__ v16h load_a_f32(const float* base, int stride, int lane) {
  int m  = lane & 15;
  int kb = (lane & 16) >> 1;
  const float* p = base + m * stride + kb;
  v4f a0 = *(const v4f*)(p);
  v4f a1 = *(const v4f*)(p + 4);
  v4f a2 = *(const v4f*)(p + 16);
  v4f a3 = *(const v4f*)(p + 20);
  v16h a;
#pragma unroll
  for (int i = 0; i < 4; ++i) {
    a[i]      = (_Float16)a0[i];
    a[i + 4]  = (_Float16)a1[i];
    a[i + 8]  = (_Float16)a2[i];
    a[i + 12] = (_Float16)a3[i];
  }
  return a;
}

// B fragment (32x16 f16) from B^T stored row-major [N][K], stride in halves.
// lane l: n = l&15; k runs [base, base+8) and [base+16, base+24), base = (l&16)>>1.
__device__ __forceinline__ v16h load_b_h(const _Float16* bt, int stride, int lane) {
  int n  = lane & 15;
  int kb = (lane & 16) >> 1;
  const _Float16* p = bt + n * stride + kb;
  v8h lo = *(const v8h*)(p);
  v8h hi = *(const v8h*)(p + 16);
  return __builtin_shufflevector(lo, hi, 0,1,2,3,4,5,6,7,8,9,10,11,12,13,14,15);
}

// ---------------- prep: weight transpose/convert + q precompute ----------------
__global__ __launch_bounds__(256) void prep_kernel(
    const float* __restrict__ attn_map, const float* __restrict__ kv_w,
    const float* __restrict__ q_w, const float* __restrict__ q_b,
    const float* __restrict__ out_w, const float* __restrict__ pol_w,
    _Float16* __restrict__ ws) {
  const int total = WS_TOTAL;
  for (int i = blockIdx.x * blockDim.x + threadIdx.x; i < total;
       i += gridDim.x * blockDim.x) {
    if (i < 131072) {                       // kv_w^T : n in [0,512), k in [0,256)
      int n = i >> 8, k = i & 255;
      ws[WS_KVW + n * 256 + k] = (_Float16)kv_w[k * 512 + n];
    } else if (i < 196608) {                // out_w^T
      int j = i - 131072;
      int n = j >> 8, k = j & 255;
      ws[WS_OWT + n * 256 + k] = (_Float16)out_w[k * 256 + n];
    } else if (i < 217088) {                // pol_w^T, padded to 80 cols
      int j = i - 196608;
      int n = j >> 8, k = j & 255;
      float v = (n < 73) ? pol_w[k * 73 + n] : 0.0f;
      ws[WS_POLT + n * 256 + k] = (_Float16)v;
    } else {                                // q = (attn_map@q_w + q_b) * 1/sqrt(32)
      int j = i - 217088;
      int s = j >> 8, c = j & 255;
      float acc = q_b[c];
      for (int t = 0; t < 64; ++t) acc += attn_map[s * 64 + t] * q_w[t * 256 + c];
      ws[WS_Q + s * 256 + c] = (_Float16)(acc * 0.17677669529663687f);
    }
  }
}

// ---------------- fused main kernel: one block per batch element ----------------
__global__ __launch_bounds__(256, 1) void fused_kernel(
    const float* __restrict__ x, const float* __restrict__ kv_b,
    const float* __restrict__ out_b, const float* __restrict__ ln_g,
    const float* __restrict__ ln_b, const float* __restrict__ pol_b,
    const _Float16* __restrict__ ws, float* __restrict__ out) {
  // 128 KB LDS, regions with overlapped lifetimes:
  //  halves [0,16384)      : K  [8][64][32]        (phase 1-2)
  //  halves [16384,32768)  : V^T[8][32][64]        (phase 1-2)
  //  bytes  [65536,131072) : x staging f32 (phase 1, TDM target) ; then
  //                          attn [8][64][64]      (phase 2) ; low half reused as
  //                          ao [64][256]          (phase 2b-3)
  //  halves [49152,65536)  : y  [64][256] f16      (phase 4-5, over attn heads 4-7)
  //  floats [0,16384)      : r  [64][256] f32      (phase 3-4, over K/V)
  __shared__ _Float16 smem[65536] __attribute__((aligned(32)));
  _Float16* kls  = smem;
  _Float16* vls  = smem + 16384;
  _Float16* attn = smem + 32768;
  _Float16* ao   = smem + 32768;
  _Float16* yls  = smem + 49152;
  float*    rls  = (float*)smem;
  float*    xls  = (float*)(smem + 32768);   // staged x[b], 64 KB, phase 1 only
  const unsigned XLS_BYTE_OFF = 65536u;      // LDS byte offset of staging region

  const _Float16* kvwT = ws + WS_KVW;
  const _Float16* owT  = ws + WS_OWT;
  const _Float16* polT = ws + WS_POLT;
  const _Float16* qws  = ws + WS_Q;

  const int b    = blockIdx.x;
  const int tid  = threadIdx.x;
  const int w    = tid >> 5;
  const int lane = tid & 31;
  const int hi8  = (lane & 16) >> 1;   // 0 or 8 (C-frag row offset)
  const int ln   = lane & 15;          // C-frag / B-frag column

  const float* xb = x + (size_t)b * (64 * 256);

  // ---------- Phase 0: TDM-stage x[b] (64x256 f32 = 64 KB) into LDS ----------
  {
    unsigned long long ga = (unsigned long long)(size_t)xb;
    v4u g0;
    g0[0] = 1u;                                   // count=1 (valid user D#)
    g0[1] = XLS_BYTE_OFF;                         // lds_addr (bytes)
    g0[2] = (unsigned)(ga & 0xFFFFFFFFu);         // global_addr[31:0]
    g0[3] = (unsigned)((ga >> 32) & 0x1FFFFFFu)   // global_addr[56:32]
            | (2u << 30);                         // type = 2 ("image")
    v8i g1;
    g1[0] = 0x00020000;                           // wg_mask=0, data_size=2 (4B)
    g1[1] = (int)(16384u << 16);                  // tensor_dim0 = 16384 (lo16)
    g1[2] = 0x00010000;                           // dim0 hi=0, tensor_dim1 = 1
    g1[3] = (int)(16384u << 16);                  // dim1 hi=0, tile_dim0 = 16384
    g1[4] = 0;                                    // tile_dim1=0, tile_dim2=0
    g1[5] = 16384;                                // tensor_dim0_stride lo32
    g1[6] = 0;
    g1[7] = 0;
    v4i gz4 = {0, 0, 0, 0};
    v8i gz8 = {0, 0, 0, 0, 0, 0, 0, 0};
    if (w == 0) {
      __builtin_amdgcn_tensor_load_to_lds(g0, g1, gz4, gz4, gz8, 0);
      __builtin_amdgcn_s_wait_tensorcnt(0);
    }
  }
  __syncthreads();
  // Warm caches for the phase-3 residual re-read (256 threads x 256B = full tile).
  __builtin_prefetch(xb + tid * 64, 0, 1);

  // ---------- Phase 1: kv = x @ kv_w + kv_b ; K -> kls, V^T -> vls ----------
  for (int mt = 0; mt < 4; ++mt) {
    v8f acc[4] = {};
    for (int ks = 0; ks < 8; ++ks) {
      v16h a = load_a_f32(xls + mt * 16 * 256 + ks * 32, 256, lane);
#pragma unroll
      for (int t = 0; t < 4; ++t) {
        int nt = w * 4 + t;
        v16h bm = load_b_h(kvwT + nt * 16 * 256 + ks * 32, 256, lane);
        acc[t] = WMMA_F16(a, bm, acc[t]);
      }
    }
#pragma unroll
    for (int t = 0; t < 4; ++t) {
      int  nt  = w * 4 + t;
      int  c0  = nt * 16;
      int  h   = c0 >> 6;
      int  rem = c0 & 63;
      bool isV = rem >= 32;
      int  d   = (rem & 31) + ln;
      float bias = kv_b[c0 + ln];
#pragma unroll
      for (int j = 0; j < 8; ++j) {
        int   m  = mt * 16 + j + hi8;
        float vv = acc[t][j] + bias;
        if (!isV) kls[h * 2048 + m * 32 + d] = (_Float16)vv;
        else      vls[h * 2048 + d * 64 + m] = (_Float16)vv;
      }
    }
  }
  __syncthreads();   // staged x dead from here; region becomes attn/ao

  // ---------- Phase 2: per-head attention (wave w <-> head w) ----------
  {
    const int h = w;
    for (int mt = 0; mt < 4; ++mt) {
      v16h aq = load_a_h(qws + mt * 16 * 256 + h * 32, 256, lane);
      v8f s[4] = {};
#pragma unroll
      for (int nt = 0; nt < 4; ++nt) {   // scores: single K=32 WMMA per tile
        v16h bm = load_b_h(kls + h * 2048 + nt * 16 * 32, 32, lane);
        s[nt] = WMMA_F16(aq, bm, s[nt]);
      }
      // softmax over the 64 keys of each row (row = VGPR j across 16-lane half)
#pragma unroll
      for (int j = 0; j < 8; ++j) {
        float mx = fmaxf(fmaxf(s[0][j], s[1][j]), fmaxf(s[2][j], s[3][j]));
        mx = fmaxf(mx, __shfl_xor(mx, 1, 16));
        mx = fmaxf(mx, __shfl_xor(mx, 2, 16));
        mx = fmaxf(mx, __shfl_xor(mx, 4, 16));
        mx = fmaxf(mx, __shfl_xor(mx, 8, 16));
        float p0 = __expf(s[0][j] - mx), p1 = __expf(s[1][j] - mx);
        float p2 = __expf(s[2][j] - mx), p3 = __expf(s[3][j] - mx);
        float sm = p0 + p1 + p2 + p3;
        sm += __shfl_xor(sm, 1, 16);
        sm += __shfl_xor(sm, 2, 16);
        sm += __shfl_xor(sm, 4, 16);
        sm += __shfl_xor(sm, 8, 16);
        float inv = 1.0f / sm;
        int m = mt * 16 + j + hi8;
        attn[h * 4096 + m * 64 +  0 + ln] = (_Float16)(p0 * inv);
        attn[h * 4096 + m * 64 + 16 + ln] = (_Float16)(p1 * inv);
        attn[h * 4096 + m * 64 + 32 + ln] = (_Float16)(p2 * inv);
        attn[h * 4096 + m * 64 + 48 + ln] = (_Float16)(p3 * inv);
      }
    }
    // out_h = attn_h @ v_h  (M=64, K=64, N=32)
    v8f o[4][2] = {};
    for (int mt = 0; mt < 4; ++mt) {
#pragma unroll
      for (int ks = 0; ks < 2; ++ks) {
        v16h a = load_a_h(attn + h * 4096 + mt * 16 * 64 + ks * 32, 64, lane);
#pragma unroll
        for (int nt = 0; nt < 2; ++nt) {
          v16h bm = load_b_h(vls + h * 2048 + nt * 16 * 64 + ks * 32, 64, lane);
          o[mt][nt] = WMMA_F16(a, bm, o[mt][nt]);
        }
      }
    }
    __syncthreads();  // all attn reads done before ao overlays it
    for (int mt = 0; mt < 4; ++mt)
#pragma unroll
      for (int nt = 0; nt < 2; ++nt)
#pragma unroll
        for (int j = 0; j < 8; ++j) {
          int m = mt * 16 + j + hi8;
          ao[m * 256 + h * 32 + nt * 16 + ln] = (_Float16)o[mt][nt][j];
        }
  }
  __syncthreads();

  // ---------- Phase 3: out-proj + bias + residual -> r (f32 LDS) ----------
  for (int mt = 0; mt < 4; ++mt) {
    v8f p[2] = {};
    for (int ks = 0; ks < 8; ++ks) {
      v16h a = load_a_h(ao + mt * 16 * 256 + ks * 32, 256, lane);
#pragma unroll
      for (int t = 0; t < 2; ++t) {
        int nt = w * 2 + t;
        v16h bm = load_b_h(owT + nt * 16 * 256 + ks * 32, 256, lane);
        p[t] = WMMA_F16(a, bm, p[t]);
      }
    }
#pragma unroll
    for (int t = 0; t < 2; ++t) {
      int   n    = (w * 2 + t) * 16 + ln;
      float bias = out_b[n];
#pragma unroll
      for (int j = 0; j < 8; ++j) {
        int m = mt * 16 + j + hi8;
        rls[m * 256 + n] = p[t][j] + bias + xb[m * 256 + n];
      }
    }
  }
  __syncthreads();

  // ---------- Phase 4: LayerNorm + ReLU -> y (f16) ----------
  {
    int row  = tid >> 2;
    int part = tid & 3;
    const float* rr = rls + row * 256 + part * 64;
    float s = 0.0f, sq = 0.0f;
    for (int i = 0; i < 64; ++i) { float v = rr[i]; s += v; sq += v * v; }
    s  += __shfl_xor(s, 1, 4);  s  += __shfl_xor(s, 2, 4);
    sq += __shfl_xor(sq, 1, 4); sq += __shfl_xor(sq, 2, 4);
    float mean = s * (1.0f / 256.0f);
    float var  = sq * (1.0f / 256.0f) - mean * mean;
    float rstd = rsqrtf(var + 1e-5f);
    for (int i = 0; i < 64; ++i) {
      int   c = part * 64 + i;
      float v = (rls[row * 256 + c] - mean) * rstd * ln_g[c] + ln_b[c];
      yls[row * 256 + c] = (_Float16)fmaxf(v, 0.0f);
    }
  }
  __syncthreads();

  // ---------- Phase 5: policy head, pol_w padded to 80 cols ----------
  for (int t = w; t < 20; t += 8) {
    int mt = t / 5, nt = t % 5;
    v8f acc = {};
    for (int ks = 0; ks < 8; ++ks) {
      v16h a  = load_a_h(yls + mt * 16 * 256 + ks * 32, 256, lane);
      v16h bm = load_b_h(polT + nt * 16 * 256 + ks * 32, 256, lane);
      acc = WMMA_F16(a, bm, acc);
    }
    int p = nt * 16 + ln;
    if (p < 73) {
      float bias = pol_b[p];
#pragma unroll
      for (int j = 0; j < 8; ++j) {
        int m = mt * 16 + j + hi8;
        out[(size_t)b * (64 * 73) + m * 73 + p] = acc[j] + bias;
      }
    }
  }
}

extern "C" void kernel_launch(void* const* d_in, const int* in_sizes, int n_in,
                              void* d_out, int out_size, void* d_ws, size_t ws_size,
                              hipStream_t stream) {
  const float* x        = (const float*)d_in[0];
  const float* attn_map = (const float*)d_in[1];
  const float* kv_w     = (const float*)d_in[2];
  const float* kv_b     = (const float*)d_in[3];
  const float* q_w      = (const float*)d_in[4];
  const float* q_b      = (const float*)d_in[5];
  const float* out_w    = (const float*)d_in[6];
  const float* out_b    = (const float*)d_in[7];
  const float* ln_g     = (const float*)d_in[8];
  const float* ln_b     = (const float*)d_in[9];
  const float* pol_w    = (const float*)d_in[10];
  const float* pol_b    = (const float*)d_in[11];
  float* out            = (float*)d_out;
  _Float16* ws          = (_Float16*)d_ws;

  prep_kernel<<<912, 256, 0, stream>>>(attn_map, kv_w, q_w, q_b, out_w, pol_w, ws);
  fused_kernel<<<4096, 256, 0, stream>>>(x, kv_b, out_b, ln_g, ln_b, pol_b, ws, out);
}
